// lowrankPhyHGalerkinConv_32650341384536
// MI455X (gfx1250) — matrix operand order repaired and checked by reading, not compiled, not run on hardware
//
#include <hip/hip_runtime.h>

// Sizes from the reference
#define BSZ   8
#define NPTS  16384
#define DIM   128
#define LMODE 128   // MODE_IN
#define KOUT  128   // MODE_OUT
#define JMODE 64    // KMODE
#define RNK   16
#define M1S   64
#define M2S   64

typedef __attribute__((ext_vector_type(16))) __bf16       v16bf;
typedef __attribute__((ext_vector_type(8)))  float        v8f;
typedef __attribute__((ext_vector_type(4)))  float        f4;
typedef __attribute__((ext_vector_type(4)))  unsigned int u4;

union FragU { u4 q[2]; v16bf v; };

// ---- f32 -> bf16 round-to-nearest-even helpers ----
__device__ __forceinline__ unsigned short f2bf(float f) {
  unsigned u = __float_as_uint(f);
  u += 0x7FFFu + ((u >> 16) & 1u);
  return (unsigned short)(u >> 16);
}
__device__ __forceinline__ unsigned pk2(float a, float b) {
  unsigned ua = __float_as_uint(a); ua += 0x7FFFu + ((ua >> 16) & 1u);
  unsigned ub = __float_as_uint(b); ub += 0x7FFFu + ((ub >> 16) & 1u);
  return (ua >> 16) | (ub & 0xFFFF0000u);
}
__device__ __forceinline__ u4 pack8(f4 a, f4 b) {
  u4 r;
  r.x = pk2(a.x, a.y); r.y = pk2(a.z, a.w);
  r.z = pk2(b.x, b.y); r.w = pk2(b.z, b.w);
  return r;
}
__device__ __forceinline__ v8f zero8() {
  v8f z;
#pragma unroll
  for (int i = 0; i < 8; ++i) z[i] = 0.0f;
  return z;
}
__device__ __forceinline__ v8f wmma_bf16(v16bf a, v16bf b, v8f c) {
  return __builtin_amdgcn_wmma_f32_16x16x32_bf16(
      /*neg_a=*/false, a, /*neg_b=*/false, b,
      /*c_mod=*/(short)0, c, /*reuse_a=*/false, /*reuse_b=*/false);
}

// A fragment (16x32 bf16) from a row-major f32 strip (row stride in floats).
// Layout: g=0 lanes -> K 0..15 ; g=1 lanes -> K 16..31 (matches B fragments).
__device__ __forceinline__ v16bf a_frag_g(const float* strip, int rstride) {
  const int lane = threadIdx.x & 31;
  const int m = lane & 15, g = lane >> 4;
  const float* p = strip + (size_t)m * rstride + 16 * g;
  f4 f0 = *(const f4*)(p);
  f4 f1 = *(const f4*)(p + 4);
  f4 f2 = *(const f4*)(p + 8);
  f4 f3 = *(const f4*)(p + 12);
  FragU u; u.q[0] = pack8(f0, f1); u.q[1] = pack8(f2, f3);
  return u.v;
}

// Raw (already bf16-packed) B fragment pointer math for LDS tiles [n][k].
__device__ __forceinline__ const unsigned short* b_frag_ptr(
    const unsigned short* base, int stride) {
  const int lane = threadIdx.x & 31;
  const int n = lane & 15, g = lane >> 4;
  return base + n * stride + 16 * g;
}

// ---------------- K0: zero split-K accumulators ----------------
__global__ __launch_bounds__(256) void k0_zero(float* p, int n) {
  int i = blockIdx.x * 256 + threadIdx.x;
  if (i < n) p[i] = 0.0f;
}

// ---------------- K1: x_hat[b,c,k] = sum_x x[b,c,x] * wbases[b,x,k] --------
// Split-K GEMM: grid (32 K-chunks, 8 b). Each WG: M=128 x N=128, K-chunk 512.
// Double-buffered LDS transpose staging; B frags preloaded before WMMA chain.
__global__ __launch_bounds__(256) void k1_xhat(const float* __restrict__ x,
                                               const float* __restrict__ wb,
                                               float* __restrict__ xhat) {
  __shared__ unsigned short Bxt[2][128 * 40];  // [n=L][k=x_pt], stride 40
  const int b = blockIdx.y;
  const int kk0 = blockIdx.x * 512;
  const int t = threadIdx.x, w = t >> 5;
  const float* xb = x + (size_t)b * DIM * NPTS;
  const float* wbb = wb + (size_t)b * NPTS * LMODE;

  auto stage = [&](int s, int buf) {
    const int r = t >> 3;         // 0..31 rows (x points)
    const int cs = (t & 7) * 16;  // 16 L-cols per thread
    const float* src = wbb + (size_t)(kk0 + s * 32 + r) * LMODE + cs;
    __builtin_prefetch(src + 32 * LMODE, 0, 1);
    f4 v0 = *(const f4*)(src);
    f4 v1 = *(const f4*)(src + 4);
    f4 v2 = *(const f4*)(src + 8);
    f4 v3 = *(const f4*)(src + 12);
    f4 vv[4] = {v0, v1, v2, v3};
#pragma unroll
    for (int u = 0; u < 16; ++u)
      Bxt[buf][(cs + u) * 40 + r] = f2bf(vv[u >> 2][u & 3]);
  };

  v8f acc[8];
#pragma unroll
  for (int nt = 0; nt < 8; ++nt) acc[nt] = zero8();

  stage(0, 0);
#pragma unroll 1
  for (int s = 0; s < 16; ++s) {
    __syncthreads();
    if (s + 1 < 16) stage(s + 1, (s + 1) & 1);
    const int kk = kk0 + s * 32;
    const v16bf a = a_frag_g(xb + (size_t)(16 * w) * NPTS + kk, NPTS);
    u4 bq[8][2];
#pragma unroll
    for (int nt = 0; nt < 8; ++nt) {
      const unsigned short* p = b_frag_ptr(&Bxt[s & 1][(16 * nt) * 40], 40);
      bq[nt][0] = *(const u4*)(p);
      bq[nt][1] = *(const u4*)(p + 8);
    }
#pragma unroll
    for (int nt = 0; nt < 8; ++nt) {
      FragU u; u.q[0] = bq[nt][0]; u.q[1] = bq[nt][1];
      acc[nt] = wmma_bf16(a, u.v, acc[nt]);
    }
  }
  const int lane = t & 31, n0 = lane & 15, g = lane >> 4;
#pragma unroll
  for (int nt = 0; nt < 8; ++nt)
#pragma unroll
    for (int v = 0; v < 8; ++v) {
      const int m = 16 * w + 8 * g + v;      // c
      const int n = 16 * nt + n0;            // L
      unsafeAtomicAdd(&xhat[(size_t)b * DIM * LMODE + (size_t)m * LMODE + n],
                      acc[nt][v]);
    }
}

// ---- K2: per (b,j): build H in LDS (bf16), x_hat1 = x_hat * H^T, store bf16
__global__ __launch_bounds__(256) void k2_xhat1(
    const float* __restrict__ xhat, const float* __restrict__ product,
    const float* __restrict__ Dout, const float* __restrict__ Din,
    const float* __restrict__ Amat, const float* __restrict__ Bmat,
    unsigned short* __restrict__ X1) {
  __shared__ unsigned short Hl[128 * 128];  // [k][l] bf16  (B operand)
  __shared__ float Qbuf[64 * 64];
  __shared__ float Din_s[128], Dout_s[128];
  const int j = blockIdx.x, b = blockIdx.y;
  const int t = threadIdx.x, w = t >> 5;

  if (t < 128) {
    Din_s[t] = Din[j * LMODE + t];
    Dout_s[t] = Dout[j * KOUT + t];
  }
  // Q[k,l] = sum_r A[j,r,k] * B[j,r,l]   (rank-16 correction)
#pragma unroll 1
  for (int it = 0; it < 16; ++it) {
    const int e = it * 256 + t;
    const int k = e >> 6, l = e & 63;
    float q = 0.0f;
#pragma unroll
    for (int r = 0; r < RNK; ++r)
      q += Amat[(size_t)j * (RNK * M1S) + r * M1S + k] *
           Bmat[(size_t)j * (RNK * M2S) + r * M2S + l];
    Qbuf[e] = q;
  }
  __syncthreads();
  // H[k,l] = Dout[j,k]*Din[j,l]*product[b,k,l]  (+ Q on the [0:64,0:64] block)
  const float* pb = product + (size_t)b * KOUT * LMODE;
#pragma unroll 1
  for (int it = 0; it < 8; ++it) {
    const int idx = it * 256 + t;
    const int k = idx >> 4, ls = (idx & 15) * 8;
    f4 p0 = *(const f4*)(pb + (size_t)k * LMODE + ls);
    f4 p1 = *(const f4*)(pb + (size_t)k * LMODE + ls + 4);
    const float dk = Dout_s[k];
    float vv[8];
#pragma unroll
    for (int u = 0; u < 8; ++u) {
      float pv = (u < 4) ? p0[u] : p1[u - 4];
      float val = dk * Din_s[ls + u] * pv;
      if (k < 64 && ls < 64) val += Qbuf[k * 64 + ls + u];
      vv[u] = val;
    }
    u4 pk;
    pk.x = pk2(vv[0], vv[1]); pk.y = pk2(vv[2], vv[3]);
    pk.z = pk2(vv[4], vv[5]); pk.w = pk2(vv[6], vv[7]);
    *(u4*)&Hl[k * 128 + ls] = pk;
  }
  __syncthreads();
  // GEMM: [128 i x 128 l] * H^T -> [128 i x 128 k]
  const float* xh = xhat + (size_t)b * DIM * LMODE;
  v8f acc[8];
#pragma unroll
  for (int nt = 0; nt < 8; ++nt) acc[nt] = zero8();
#pragma unroll 1
  for (int l0 = 0; l0 < 128; l0 += 32) {
    const v16bf a = a_frag_g(xh + (size_t)(16 * w) * LMODE + l0, LMODE);
    u4 bq[8][2];
#pragma unroll
    for (int nt = 0; nt < 8; ++nt) {
      const unsigned short* p = b_frag_ptr(&Hl[(16 * nt) * 128 + l0], 128);
      bq[nt][0] = *(const u4*)(p);
      bq[nt][1] = *(const u4*)(p + 8);
    }
#pragma unroll
    for (int nt = 0; nt < 8; ++nt) {
      FragU u; u.q[0] = bq[nt][0]; u.q[1] = bq[nt][1];
      acc[nt] = wmma_bf16(a, u.v, acc[nt]);
    }
  }
  // store x_hat1 as bf16, layout [b][(i*64 + j)][k]
  const int lane = t & 31, n0 = lane & 15, g = lane >> 4;
#pragma unroll
  for (int nt = 0; nt < 8; ++nt)
#pragma unroll
    for (int v = 0; v < 8; ++v) {
      const int iRow = 16 * w + 8 * g + v;
      const int kcol = 16 * nt + n0;
      X1[((size_t)b * (DIM * JMODE) + (size_t)iRow * JMODE + j) * KOUT + kcol] =
          f2bf(acc[nt][v]);
    }
}

// ---- K3: y[b,o,k] = sum_{c=(i,j)} weights[i,o,j] * X1[b,c,k]  (split-K) ----
__global__ __launch_bounds__(256) void k3_y(const float* __restrict__ weights,
                                            const unsigned short* __restrict__ X1,
                                            float* __restrict__ yws) {
  __shared__ unsigned short Xt[2][128 * 40];  // [n=k][K=c], stride 40
  const int b = blockIdx.y, chunk = blockIdx.x;
  const int t = threadIdx.x, w = t >> 5;

  auto stage = [&](int s, int buf) {
    const int c0 = chunk * 512 + s * 32;
    const int cl = t >> 3;            // 0..31 local c
    const int ks = (t & 7) * 16;      // 16 k-cols
    const unsigned short* src =
        X1 + ((size_t)b * (DIM * JMODE) + (size_t)(c0 + cl)) * KOUT + ks;
    __builtin_prefetch(src + 32 * KOUT, 0, 1);
    u4 a0 = *(const u4*)(src);
    u4 a1 = *(const u4*)(src + 8);
    unsigned wv[8] = {a0.x, a0.y, a0.z, a0.w, a1.x, a1.y, a1.z, a1.w};
#pragma unroll
    for (int u = 0; u < 16; ++u) {
      unsigned short h =
          (unsigned short)((u & 1) ? (wv[u >> 1] >> 16) : (wv[u >> 1] & 0xFFFFu));
      Xt[buf][(ks + u) * 40 + cl] = h;
    }
  };

  v8f acc[8];
#pragma unroll
  for (int nt = 0; nt < 8; ++nt) acc[nt] = zero8();

  stage(0, 0);
#pragma unroll 1
  for (int s = 0; s < 16; ++s) {
    __syncthreads();
    if (s + 1 < 16) stage(s + 1, (s + 1) & 1);
    const int c0 = chunk * 512 + s * 32;   // 32 | 64 so (i, j0) constant in tile
    const int i = c0 >> 6, j0 = c0 & 63;
    // A operand: W2[o, c] = weights[i, o, j] -> contiguous over j
    const v16bf a =
        a_frag_g(weights + (size_t)i * (DIM * JMODE) + (size_t)(16 * w) * JMODE + j0,
                 JMODE);
    u4 bq[8][2];
#pragma unroll
    for (int nt = 0; nt < 8; ++nt) {
      const unsigned short* p = b_frag_ptr(&Xt[s & 1][(16 * nt) * 40], 40);
      bq[nt][0] = *(const u4*)(p);
      bq[nt][1] = *(const u4*)(p + 8);
    }
#pragma unroll
    for (int nt = 0; nt < 8; ++nt) {
      FragU u; u.q[0] = bq[nt][0]; u.q[1] = bq[nt][1];
      acc[nt] = wmma_bf16(a, u.v, acc[nt]);
    }
  }
  const int lane = t & 31, n0 = lane & 15, g = lane >> 4;
#pragma unroll
  for (int nt = 0; nt < 8; ++nt)
#pragma unroll
    for (int v = 0; v < 8; ++v) {
      const int m = 16 * w + 8 * g + v;  // o
      const int n = 16 * nt + n0;        // k
      unsafeAtomicAdd(&yws[(size_t)b * DIM * KOUT + (size_t)m * KOUT + n],
                      acc[nt][v]);
    }
}

// ---- K4: out[b,c,x] = sum_k y[b,c,k] * bases[x,k] ----
// bases tile converted once per WG into LDS (k contiguous, no transpose),
// shared by all 8 waves; A frags (y rows) are wave-exclusive -> from global.
__global__ __launch_bounds__(256) void k4_out(const float* __restrict__ yws,
                                              const float* __restrict__ bases,
                                              float* __restrict__ out) {
  __shared__ unsigned short Bas[128 * 128];  // [x_local][k] bf16, 32 KB
  const int xt0 = blockIdx.x * 128;
  const int b = blockIdx.y;
  const int t = threadIdx.x, w = t >> 5;

  // stage bases[xt0 .. xt0+127][0..127] -> bf16 LDS
#pragma unroll 1
  for (int it = 0; it < 8; ++it) {
    const int idx = it * 256 + t;
    const int row = idx >> 4, ks = (idx & 15) * 8;
    const float* src = bases + (size_t)(xt0 + row) * KOUT + ks;
    f4 p0 = *(const f4*)(src);
    f4 p1 = *(const f4*)(src + 4);
    u4 pk;
    pk.x = pk2(p0.x, p0.y); pk.y = pk2(p0.z, p0.w);
    pk.z = pk2(p1.x, p1.y); pk.w = pk2(p1.z, p1.w);
    *(u4*)&Bas[row * 128 + ks] = pk;
  }
  __syncthreads();

  const float* yb = yws + (size_t)b * DIM * KOUT;
  v8f acc[8];
#pragma unroll
  for (int nt = 0; nt < 8; ++nt) acc[nt] = zero8();
#pragma unroll 1
  for (int k0 = 0; k0 < 128; k0 += 32) {
    const v16bf a = a_frag_g(yb + (size_t)(16 * w) * KOUT + k0, KOUT);
    u4 bq[8][2];
#pragma unroll
    for (int nt = 0; nt < 8; ++nt) {
      const unsigned short* p = b_frag_ptr(&Bas[(16 * nt) * 128 + k0], 128);
      bq[nt][0] = *(const u4*)(p);
      bq[nt][1] = *(const u4*)(p + 8);
    }
#pragma unroll
    for (int nt = 0; nt < 8; ++nt) {
      FragU u; u.q[0] = bq[nt][0]; u.q[1] = bq[nt][1];
      acc[nt] = wmma_bf16(a, u.v, acc[nt]);
    }
  }
  const int lane = t & 31, n0 = lane & 15, g = lane >> 4;
#pragma unroll
  for (int nt = 0; nt < 8; ++nt)
#pragma unroll
    for (int v = 0; v < 8; ++v) {
      const int c = 16 * w + 8 * g + v;
      const int xg = xt0 + 16 * nt + n0;
      out[(size_t)b * DIM * NPTS + (size_t)c * NPTS + xg] = acc[nt][v];
    }
}

extern "C" void kernel_launch(void* const* d_in, const int* in_sizes, int n_in,
                              void* d_out, int out_size, void* d_ws, size_t ws_size,
                              hipStream_t stream) {
  const float* x       = (const float*)d_in[0];
  const float* wbases  = (const float*)d_in[1];
  const float* bases   = (const float*)d_in[2];
  const float* product = (const float*)d_in[3];
  const float* Dout    = (const float*)d_in[4];
  const float* Din     = (const float*)d_in[5];
  const float* Amat    = (const float*)d_in[6];
  const float* Bmat    = (const float*)d_in[7];
  const float* weights = (const float*)d_in[8];
  float* out = (float*)d_out;

  char* ws = (char*)d_ws;
  float* xhat = (float*)ws;                                  // 512 KB
  float* yws  = (float*)(ws + 524288);                       // 512 KB
  unsigned short* X1 = (unsigned short*)(ws + 1048576);      // 16 MB bf16

  // zero the two split-K accumulators (first 262144 floats of ws)
  k0_zero<<<1024, 256, 0, stream>>>((float*)ws, 262144);
  k1_xhat<<<dim3(32, 8), 256, 0, stream>>>(x, wbases, xhat);
  k2_xhat1<<<dim3(64, 8), 256, 0, stream>>>(xhat, product, Dout, Din, Amat, Bmat, X1);
  k3_y<<<dim3(16, 8), 256, 0, stream>>>(weights, X1, yws);
  k4_out<<<dim3(128, 8), 256, 0, stream>>>(yws, bases, out);
}
